// GymAgent_29051158790176
// MI455X (gfx1250) — compile-verified, matrix operand order checked by reading, not run on hardware
//
#include <hip/hip_runtime.h>

typedef __attribute__((ext_vector_type(16))) _Float16 v16h;
typedef __attribute__((ext_vector_type(8)))  _Float16 v8h;
typedef __attribute__((ext_vector_type(8)))  float    v8f;

// ---------------- WMMA helpers (CDNA5 wave32, 16x16x32 f16 -> f32) ----------------

static __device__ __forceinline__ v8f wmma16(v16h a, v16h b, v8f c) {
  // (neg_a, A, neg_b, B, c_mod, C, reuse_a, reuse_b)
  return __builtin_amdgcn_wmma_f32_16x16x32_f16(false, a, false, b, (short)0, c,
                                                false, false);
}

static __device__ __forceinline__ v8f zero8() {
  v8f z;
#pragma unroll
  for (int i = 0; i < 8; ++i) z[i] = 0.f;
  return z;
}

static __device__ __forceinline__ float sigm(float x) {
  return 1.f / (1.f + __expf(-x));
}

// A-fragment (16x32 f16), ISA 7.12.2: lane L holds row m=L%16; half = L/16.
// elem i   <-> K = half*8 + i          (i in 0..7)
// elem 8+i <-> K = 16 + half*8 + i
static __device__ __forceinline__ v16h fragA(const _Float16* base, int lda, int lane) {
  const _Float16* p = base + (lane & 15) * lda + ((lane >> 4) << 3);
  v8h lo = *(const v8h*)p;
  v8h hi = *(const v8h*)(p + 16);
  v16h a;
#pragma unroll
  for (int i = 0; i < 8; ++i) { a[i] = lo[i]; a[8 + i] = hi[i]; }
  return a;
}

// Pre-packed B fragment: tile-major / lane-major / 16 contiguous halves.
// One aligned 32B load per lane, zero packing ALU.
static __device__ __forceinline__ v16h fragBpk(const _Float16* __restrict__ pk,
                                               int tile, int lane) {
  return *(const v16h*)(pk + ((((size_t)tile << 5) + (size_t)lane) << 4));
}

// Async stage of 32 bytes (16 halves) from global to LDS via CDNA5 async copy.
// VDST carries the LDS byte offset (flat LDS addr[31:0] == LDS offset, ISA 10.2).
static __device__ __forceinline__ void async_copy32(const _Float16* g, unsigned lds) {
  asm volatile(
      "global_load_async_to_lds_b128 %0, %2, off\n\t"
      "global_load_async_to_lds_b128 %1, %3, off"
      :
      : "v"(lds), "v"(lds + 16u), "v"(g), "v"(g + 8)
      : "memory");
}

static __device__ __forceinline__ void async_wait0() {
  asm volatile("s_wait_asynccnt 0" ::: "memory");
}

// -------- weight repack into WMMA B-fragment order --------
// src = torch weight (N, K) row-major; dst element order:
// idx = ((kt*(N/16) + nt)*32 + lane)*16 + i ;  n = nt*16 + lane%16,
// k = kt*32 + (i<8 ? (lane/16)*8 + i : 16 + (lane/16)*8 + (i-8))
__global__ void packB_kernel(const float* __restrict__ src,
                             _Float16* __restrict__ dst, int N, int K) {
  int idx = blockIdx.x * 256 + threadIdx.x;
  if (idx >= N * K) return;
  int i    = idx & 15;
  int lane = (idx >> 4) & 31;
  int tile = idx >> 9;
  int ntiles = N >> 4;
  int kt = tile / ntiles, nt = tile - kt * ntiles;
  int n  = nt * 16 + (lane & 15);
  int h8 = (lane >> 4) << 3;
  int k  = kt * 32 + ((i < 8) ? (h8 + i) : (16 + h8 + (i - 8)));
  dst[idx] = (_Float16)src[(size_t)n * K + k];
}

// ---------------- fused conv stack: one image per workgroup ----------------
// conv1: 1x84x84 --8x8/s4--> 32x20x20  (M=400,K=64, N=32)
// conv2: 32x20x20 --4x4/s2--> 64x9x9   (M=81, K=512,N=64)
// conv3: 64x9x9  --3x3/s1--> 64x7x7    (M=49, K=576,N=64)
// output row (3136) in torch flatten order c*49 + oy*7 + ox, f16.

__global__ __launch_bounds__(256) void conv_stack_kernel(
    const float* __restrict__ x,
    const _Float16* __restrict__ B1pk, const float* __restrict__ b1,
    const _Float16* __restrict__ B2pk, const float* __restrict__ b2,
    const _Float16* __restrict__ B3pk, const float* __restrict__ b3,
    _Float16* __restrict__ conv3out) {
  __shared__ __align__(16) _Float16 s_img[84 * 88];
  __shared__ __align__(16) _Float16 s_c1[32 * 400];
  __shared__ __align__(16) _Float16 s_c2[64 * 81];

  const int img  = blockIdx.x;
  const int tid  = threadIdx.x;
  const int wave = tid >> 5, lane = tid & 31;
  const int n_lo = lane & 15;
  const int h8   = (lane >> 4) << 3;   // K half offset for A frags
  const int mo   = h8;                 // C/D row offset: 8*(lane/16)

  const float* xin = x + (size_t)img * 84 * 84;
  for (int i = tid; i < 84 * 84; i += 256) {
    int y = i / 84, xx = i - y * 84;
    s_img[y * 88 + xx] = (_Float16)(xin[i] * (1.0f / 255.0f));
  }
  __syncthreads();

  // ---- conv1 (B tiles: ks*2 + nt) ----
  for (int tile = wave; tile < 50; tile += 8) {
    int mt = tile >> 1, nt = tile & 1;
    int m0 = mt * 16, n0 = nt * 16;
    int m = m0 + (lane & 15);
    int oy = m / 20, ox = m - oy * 20;
    v8f acc = zero8();
#pragma unroll
    for (int ks = 0; ks < 2; ++ks) {
      int k0 = ks * 32;
      v16h a;
#pragma unroll
      for (int i = 0; i < 8; ++i) {
        int k  = k0 + h8 + i;
        int k2 = k + 16;
        a[i]     = s_img[(4 * oy + (k  >> 3)) * 88 + 4 * ox + (k  & 7)];
        a[8 + i] = s_img[(4 * oy + (k2 >> 3)) * 88 + 4 * ox + (k2 & 7)];
      }
      v16h b = fragBpk(B1pk, ks * 2 + nt, lane);
      acc = wmma16(a, b, acc);
    }
    float bv = b1[n0 + n_lo];
#pragma unroll
    for (int r = 0; r < 8; ++r) {
      int mm = m0 + mo + r;
      float v = fmaxf(acc[r] + bv, 0.f);
      s_c1[(n0 + n_lo) * 400 + mm] = (_Float16)v;
    }
  }
  __syncthreads();

  // ---- conv2 (B tiles: ks*4 + nt) ----
  for (int tile = wave; tile < 24; tile += 8) {
    int mt = tile >> 2, nt = tile & 3;
    int m0 = mt * 16, n0 = nt * 16;
    int m = m0 + (lane & 15); if (m > 80) m = 80;
    int oy = m / 9, ox = m - oy * 9;
    v8f acc = zero8();
    for (int ks = 0; ks < 16; ++ks) {
      int k0 = ks * 32;
      v16h a;
#pragma unroll
      for (int i = 0; i < 8; ++i) {
        int k  = k0 + h8 + i;
        int k2 = k + 16;
        int c1i = k >> 4,  r1 = k & 15;
        int c2i = k2 >> 4, r2 = k2 & 15;
        a[i]     = s_c1[c1i * 400 + (2 * oy + (r1 >> 2)) * 20 + 2 * ox + (r1 & 3)];
        a[8 + i] = s_c1[c2i * 400 + (2 * oy + (r2 >> 2)) * 20 + 2 * ox + (r2 & 3)];
      }
      v16h b = fragBpk(B2pk, ks * 4 + nt, lane);
      acc = wmma16(a, b, acc);
    }
    float bv = b2[n0 + n_lo];
#pragma unroll
    for (int r = 0; r < 8; ++r) {
      int mm = m0 + mo + r;
      if (mm < 81) {
        float v = fmaxf(acc[r] + bv, 0.f);
        s_c2[(n0 + n_lo) * 81 + mm] = (_Float16)v;
      }
    }
  }
  __syncthreads();

  // ---- conv3 (B tiles: ks*4 + nt) ----
  _Float16* outp = conv3out + (size_t)img * 3136;
  for (int tile = wave; tile < 16; tile += 8) {
    int mt = tile >> 2, nt = tile & 3;
    int m0 = mt * 16, n0 = nt * 16;
    int m = m0 + (lane & 15); if (m > 48) m = 48;
    int oy = m / 7, ox = m - oy * 7;
    v8f acc = zero8();
    for (int ks = 0; ks < 18; ++ks) {
      int k0 = ks * 32;
      v16h a;
#pragma unroll
      for (int i = 0; i < 8; ++i) {
        int k  = k0 + h8 + i;
        int k2 = k + 16;
        int c1i = k / 9,  r1 = k - c1i * 9;
        int c2i = k2 / 9, r2 = k2 - c2i * 9;
        a[i]     = s_c2[c1i * 81 + (oy + r1 / 3) * 9 + ox + (r1 % 3)];
        a[8 + i] = s_c2[c2i * 81 + (oy + r2 / 3) * 9 + ox + (r2 % 3)];
      }
      v16h b = fragBpk(B3pk, ks * 4 + nt, lane);
      acc = wmma16(a, b, acc);
    }
    float bv = b3[n0 + n_lo];
#pragma unroll
    for (int r = 0; r < 8; ++r) {
      int mm = m0 + mo + r;
      if (mm < 49) {
        float v = fmaxf(acc[r] + bv, 0.f);
        outp[(n0 + n_lo) * 49 + mm] = (_Float16)v;
      }
    }
  }
}

// ---------------- tiled WMMA GEMM (A: 4096 x K f16, B packed K x 512 f16) -------
// Block tile 128x128, 8 waves (4x2), A staged to LDS via async copy,
// B fragments loaded pre-packed straight from global (L2 resident).
// MODE: 0 = ReLU + f16 store (FC), 1 = bias2 add + f32 store (gates).

template <int MODE>
__global__ __launch_bounds__(256) void gemm512_kernel(
    const _Float16* __restrict__ A, const _Float16* __restrict__ Bpk,
    const float* __restrict__ bias0, const float* __restrict__ bias1,
    void* __restrict__ Cout, int K) {
  __shared__ __align__(16) _Float16 As[128 * 40];
  const int M0 = blockIdx.x * 128, N0 = blockIdx.y * 128;
  const int tid = threadIdx.x, wave = tid >> 5, lane = tid & 31;
  const int wr = wave & 3, wc = wave >> 2;  // 32 rows x 64 cols per wave

  v8f acc[2][4];
#pragma unroll
  for (int i = 0; i < 2; ++i)
#pragma unroll
    for (int j = 0; j < 4; ++j) acc[i][j] = zero8();

  const int arow = tid >> 1, acol = (tid & 1) << 4;
  const unsigned lds_a =
      (unsigned)(uintptr_t)&As[arow * 40 + acol];  // LDS byte offset (addr[31:0])
  const int ntile0 = (N0 >> 4) + wc * 4;           // packed-B column tile base

  for (int k0 = 0; k0 < K; k0 += 32) {
    const _Float16* ag = A + (size_t)(M0 + arow) * K + k0 + acol;
    async_copy32(ag, lds_a);                       // ASYNCcnt-tracked LDS fill
    __builtin_prefetch((const void*)(ag + 32), 0, 0);
    async_wait0();
    __syncthreads();

    v16h af[2];
#pragma unroll
    for (int i = 0; i < 2; ++i)
      af[i] = fragA(&As[(wr * 32 + i * 16) * 40], 40, lane);
    const int bt = (k0 >> 5) * 32 + ntile0;        // K-tile * (512/16) + col tile
#pragma unroll
    for (int j = 0; j < 4; ++j) {
      v16h bf = fragBpk(Bpk, bt + j, lane);
#pragma unroll
      for (int i = 0; i < 2; ++i) acc[i][j] = wmma16(af[i], bf, acc[i][j]);
    }
    __syncthreads();
  }

  const int n_lo = lane & 15, mo = (lane >> 4) << 3;
#pragma unroll
  for (int i = 0; i < 2; ++i)
#pragma unroll
    for (int j = 0; j < 4; ++j) {
      int n = N0 + wc * 64 + j * 16 + n_lo;
      float bv = (MODE == 0) ? bias0[n] : (bias0[n] + bias1[n]);
#pragma unroll
      for (int r = 0; r < 8; ++r) {
        int m = M0 + wr * 32 + i * 16 + mo + r;
        float v = acc[i][j][r] + bv;
        if (MODE == 0) {
          ((_Float16*)Cout)[(size_t)m * 512 + n] = (_Float16)fmaxf(v, 0.f);
        } else {
          ((float*)Cout)[(size_t)m * 512 + n] = v;
        }
      }
    }
}

// ---------------- sequential LSTM scan: single persistent block ----------------
// B=32, H=128, T=128. 8 waves; wave w owns hidden columns [w*16, w*16+16) and
// computes the i/f/g/o WMMA tiles for that slice -> gate nonlinearity fully in
// registers (C layout keeps i,f,g,o for one (b,j) in one lane).

__global__ __launch_bounds__(256) void lstm_scan_kernel(
    const float* __restrict__ gates,      // (T*B, 512), biases folded
    const _Float16* __restrict__ BhhPk,   // packed W_hh^T (K=128, N=512)
    const float* __restrict__ done,
    const float* __restrict__ h0, const float* __restrict__ c0,
    float* __restrict__ hs,               // (T*B, 128)
    float* __restrict__ outHC) {          // [0,4096)=hT, [4096,8192)=cT
  __shared__ __align__(16) float    sH[32 * 128];
  __shared__ __align__(16) float    sC[32 * 128];
  __shared__ __align__(16) _Float16 sH16[32 * 128];

  const int tid = threadIdx.x, wave = tid >> 5, lane = tid & 31;
  const int n_lo = lane & 15, mo = (lane >> 4) << 3;
  const int jb = wave * 16;

  for (int i = tid; i < 4096; i += 256) { sH[i] = h0[i]; sC[i] = c0[i]; }
  __syncthreads();

  for (int t = 0; t < 128; ++t) {
    for (int i = tid; i < 4096; i += 256) {
      float mk = 1.f - done[t * 32 + (i >> 7)];
      float hv = sH[i] * mk, cv = sC[i] * mk;
      sH[i] = hv; sC[i] = cv; sH16[i] = (_Float16)hv;
    }
    __syncthreads();

    v8f acc[2][4];
    const float* grow = gates + (size_t)(t * 32) * 512;
#pragma unroll
    for (int mt = 0; mt < 2; ++mt)
#pragma unroll
      for (int g = 0; g < 4; ++g) {
        v8f cv;
#pragma unroll
        for (int r = 0; r < 8; ++r)
          cv[r] = grow[(size_t)(mt * 16 + mo + r) * 512 + g * 128 + jb + n_lo];
        acc[mt][g] = cv;
      }

    // gates += h @ W_hh^T   (M=32, K=128; packed B tile = ks*32 + g*8 + wave)
#pragma unroll
    for (int ks = 0; ks < 4; ++ks) {
      int k0 = ks * 32;
      v16h af[2];
#pragma unroll
      for (int mt = 0; mt < 2; ++mt)
        af[mt] = fragA(&sH16[(mt * 16) * 128 + k0], 128, lane);
#pragma unroll
      for (int g = 0; g < 4; ++g) {
        v16h bf = fragBpk(BhhPk, ks * 32 + g * 8 + wave, lane);
#pragma unroll
        for (int mt = 0; mt < 2; ++mt) acc[mt][g] = wmma16(af[mt], bf, acc[mt][g]);
      }
    }

#pragma unroll
    for (int mt = 0; mt < 2; ++mt)
#pragma unroll
      for (int r = 0; r < 8; ++r) {
        int b = mt * 16 + mo + r;
        int j = jb + n_lo;
        float gi = acc[mt][0][r], gf = acc[mt][1][r];
        float gg = acc[mt][2][r], go = acc[mt][3][r];
        float cv = sC[b * 128 + j];
        float cn = sigm(gf) * cv + sigm(gi) * tanhf(gg);
        float hn = sigm(go) * tanhf(cn);
        sC[b * 128 + j] = cn;
        sH[b * 128 + j] = hn;
        hs[(size_t)(t * 32 + b) * 128 + j] = hn;
      }
    __syncthreads();
  }

  for (int i = tid; i < 4096; i += 256) {
    outHC[i]        = sH[i];
    outHC[4096 + i] = sC[i];
  }
}

// ---------------- heads: logits (4096x6) + value (4096) ----------------

__global__ void heads_kernel(const float* __restrict__ hs,
                             const float* __restrict__ Wa, const float* __restrict__ ba,
                             const float* __restrict__ Wc, const float* __restrict__ bc,
                             float* __restrict__ out) {
  int tid = blockIdx.x * 256 + threadIdx.x;
  if (tid >= 4096 * 7) return;
  int r = tid / 7, o = tid - r * 7;
  const float* hrow = hs + (size_t)r * 128;
  const float* w = (o < 6) ? (Wa + o * 128) : Wc;
  float s = 0.f;
#pragma unroll 4
  for (int k = 0; k < 128; ++k) s += hrow[k] * w[k];
  if (o < 6) out[(size_t)r * 6 + o] = s + ba[o];
  else       out[24576 + r]         = s + bc[0];
}

// ---------------- host launch ----------------

extern "C" void kernel_launch(void* const* d_in, const int* in_sizes, int n_in,
                              void* d_out, int out_size, void* d_ws, size_t ws_size,
                              hipStream_t stream) {
  (void)in_sizes; (void)n_in; (void)out_size; (void)ws_size;
  const float* x    = (const float*)d_in[0];
  const float* done = (const float*)d_in[1];
  const float* h0   = (const float*)d_in[2];
  const float* c0   = (const float*)d_in[3];
  const float* W1   = (const float*)d_in[4];
  const float* b1   = (const float*)d_in[5];
  const float* W2   = (const float*)d_in[6];
  const float* b2   = (const float*)d_in[7];
  const float* W3   = (const float*)d_in[8];
  const float* b3   = (const float*)d_in[9];
  const float* Wfc  = (const float*)d_in[10];
  const float* bfc  = (const float*)d_in[11];
  const float* Wih  = (const float*)d_in[12];
  const float* Whh  = (const float*)d_in[13];
  const float* bih  = (const float*)d_in[14];
  const float* bhh  = (const float*)d_in[15];
  const float* Wa   = (const float*)d_in[16];
  const float* ba   = (const float*)d_in[17];
  const float* Wc   = (const float*)d_in[18];
  const float* bc   = (const float*)d_in[19];
  float* out = (float*)d_out;

  char* ws = (char*)d_ws;
  size_t off = 0;
  auto carve = [&](size_t bytes) {
    void* p = ws + off;
    off = (off + bytes + 255) & ~(size_t)255;
    return p;
  };
  _Float16* B1pk  = (_Float16*)carve((size_t)64 * 32 * 2);
  _Float16* B2pk  = (_Float16*)carve((size_t)512 * 64 * 2);
  _Float16* B3pk  = (_Float16*)carve((size_t)576 * 64 * 2);
  _Float16* BfcPk = (_Float16*)carve((size_t)3136 * 512 * 2);
  _Float16* BihPk = (_Float16*)carve((size_t)512 * 512 * 2);
  _Float16* BhhPk = (_Float16*)carve((size_t)128 * 512 * 2);
  _Float16* conv3 = (_Float16*)carve((size_t)4096 * 3136 * 2);
  _Float16* hid   = (_Float16*)carve((size_t)4096 * 512 * 2);
  float*    gates = (float*)carve((size_t)4096 * 512 * 4);
  float*    hsbuf = (float*)carve((size_t)4096 * 128 * 4);

  // repack weights to f16 WMMA B-fragment layouts
  packB_kernel<<<(32 * 64 + 255) / 256, 256, 0, stream>>>(W1, B1pk, 32, 64);
  packB_kernel<<<(64 * 512 + 255) / 256, 256, 0, stream>>>(W2, B2pk, 64, 512);
  packB_kernel<<<(64 * 576 + 255) / 256, 256, 0, stream>>>(W3, B3pk, 64, 576);
  packB_kernel<<<(512 * 3136 + 255) / 256, 256, 0, stream>>>(Wfc, BfcPk, 512, 3136);
  packB_kernel<<<(512 * 512 + 255) / 256, 256, 0, stream>>>(Wih, BihPk, 512, 512);
  packB_kernel<<<(512 * 128 + 255) / 256, 256, 0, stream>>>(Whh, BhhPk, 512, 128);

  // fused conv stack, one image per workgroup
  conv_stack_kernel<<<4096, 256, 0, stream>>>(x, B1pk, b1, B2pk, b2, B3pk, b3, conv3);

  // FC: hidden = relu(conv3 @ Wfc^T + bfc)   (4096 x 3136 x 512)
  gemm512_kernel<0><<<dim3(32, 4), 256, 0, stream>>>(conv3, BfcPk, bfc, nullptr,
                                                     (void*)hid, 3136);

  // gates_x = hidden @ W_ih^T + b_ih + b_hh  (4096 x 512 x 512)
  gemm512_kernel<1><<<dim3(32, 4), 256, 0, stream>>>(hid, BihPk, bih, bhh,
                                                     (void*)gates, 512);

  // sequential LSTM scan; writes hs and hT/cT tail of output
  lstm_scan_kernel<<<1, 256, 0, stream>>>(gates, BhhPk, done, h0, c0, hsbuf,
                                          out + 28672);

  // heads
  heads_kernel<<<(4096 * 7 + 255) / 256, 256, 0, stream>>>(hsbuf, Wa, ba, Wc, bc, out);
}
